// ReformerPolicy_82746839924979
// MI455X (gfx1250) — matrix operand order repaired and checked by reading, not compiled, not run on hardware
//
#include <hip/hip_runtime.h>
#include <hip/hip_bf16.h>
#include <math.h>

// ---------------- types ----------------
typedef unsigned short u16;
typedef __attribute__((ext_vector_type(16))) __bf16       v16bf;
typedef __attribute__((ext_vector_type(8)))  float        v8f;
typedef __attribute__((ext_vector_type(4)))  unsigned int u32x4;
typedef __attribute__((ext_vector_type(4)))  int          v4i;

union ABu { v16bf v; u32x4 q[2]; };   // one WMMA A/B fragment = 32 bytes
union U4u { u32x4 q; u16 s[8]; };     // 16-byte chunk as 8 bf16

// ---- CDNA5 async global->LDS path (guarded; falls back to VGPR staging) ----
#if defined(__gfx1250__) && \
    __has_builtin(__builtin_amdgcn_global_load_async_to_lds_b128) && \
    __has_builtin(__builtin_amdgcn_s_wait_asynccnt)
#define USE_ASYNC_LDS 1
#define ASYNC_CP16(gsrc, ldst)                                             \
    __builtin_amdgcn_global_load_async_to_lds_b128(                        \
        (__attribute__((address_space(1))) v4i*)(gsrc),                    \
        (__attribute__((address_space(3))) v4i*)(ldst), 0, 0)
#define ASYNC_WAIT() __builtin_amdgcn_s_wait_asynccnt(0)
#else
#define USE_ASYNC_LDS 0
#define ASYNC_WAIT()
#endif

__device__ __forceinline__ u16 f2b(float f) {
    unsigned u = __float_as_uint(f);
    u += 0x7FFFu + ((u >> 16) & 1u);          // round-to-nearest-even
    return (u16)(u >> 16);
}
__device__ __forceinline__ float b2f(u16 h) {
    return __uint_as_float(((unsigned)h) << 16);
}
__device__ __forceinline__ v8f zero8() {
    v8f z = {0.f,0.f,0.f,0.f,0.f,0.f,0.f,0.f};
    return z;
}

// ---------------- model dims ----------------
static constexpr int Bc = 4, Tc = 4096, SDc = 256, Cc = 1024, Ac = 64;
static constexpr int Lc = 6, Hc = 16, BSc = 64, HDc = 64;
static constexpr int Mc = Bc * Tc;            // 16384 rows

// =====================================================================
// Generic bf16 GEMM:  C[M,N] = A[M,K] @ B[K,N]  (+ fused epilogues)
// Block tile 128x128, K-step 32, 256 threads (8 wave32, 4x2 wave grid),
// each wave computes 32x64 via 2x4 v_wmma_f32_16x16x32_bf16.
// Double-buffered LDS; A tile staged via async global->LDS when available.
// EPI 0: Cb = bf16(acc + bias)
// EPI 1: Cf = extra(residual) + acc + bias        (f32 out)
// EPI 2: Cb = bf16(gelu(acc + bias))              (exact erf gelu)
// EPI 3: Cf = acc + bias + extra[(row%Tm)*N+col]  (pos-embed add)
// =====================================================================
template<int EPI>
__global__ __launch_bounds__(256) void gemm_bf16(
    const u16* __restrict__ Ag, const u16* __restrict__ Bg,
    const float* __restrict__ bias, const float* extra,
    float* Cf, u16* __restrict__ Cb,
    int Mdim, int Ndim, int Kdim, int Tm)
{
    constexpr int LDA = 40, LDB = 40;               // 32 + 8 pad (bf16 elems)
    __shared__ __align__(16) u16 sA[2][128 * LDA];  // A tile  [m][k]
    __shared__ __align__(16) u16 sB[2][128 * LDB];  // B tile transposed [n][k]

    const int tid  = threadIdx.x;
    const int wid  = tid >> 5, lane = tid & 31;
    const int lr   = lane & 15, half = lane >> 4;
    const int wm   = wid & 3,  wn   = wid >> 2;     // 4 x 2 waves
    const int m0   = blockIdx.y * 128, n0 = blockIdx.x * 128;

    // staging helpers -------------------------------------------------
    auto loadB = [&](int k0, U4u (&ub)[2]) {
        #pragma unroll
        for (int s = 0; s < 2; ++s) {
            int ch = tid + s * 256;
            int kk = ch >> 4, nn = (ch & 15) << 3;
            ub[s].q = *reinterpret_cast<const u32x4*>(&Bg[(size_t)(k0 + kk) * Ndim + n0 + nn]);
        }
    };
    auto writeB = [&](int buf, U4u (&ub)[2]) {      // transpose into sB[n][k]
        #pragma unroll
        for (int s = 0; s < 2; ++s) {
            int ch = tid + s * 256;
            int kk = ch >> 4, nn = (ch & 15) << 3;
            #pragma unroll
            for (int e = 0; e < 8; ++e) sB[buf][(nn + e) * LDB + kk] = ub[s].s[e];
        }
    };
#if USE_ASYNC_LDS
    auto stageA = [&](int k0, int buf) {            // direct global->LDS DMA
        #pragma unroll
        for (int s = 0; s < 2; ++s) {
            int ch = tid + s * 256;
            int r = ch >> 2, cc = (ch & 3) << 3;
            ASYNC_CP16(&Ag[(size_t)(m0 + r) * Kdim + k0 + cc],
                       &sA[buf][r * LDA + cc]);
        }
    };
#else
    auto loadA = [&](int k0, U4u (&ua)[2]) {
        #pragma unroll
        for (int s = 0; s < 2; ++s) {
            int ch = tid + s * 256;
            int r = ch >> 2, cc = (ch & 3) << 3;
            ua[s].q = *reinterpret_cast<const u32x4*>(&Ag[(size_t)(m0 + r) * Kdim + k0 + cc]);
        }
    };
    auto writeA = [&](int buf, U4u (&ua)[2]) {
        #pragma unroll
        for (int s = 0; s < 2; ++s) {
            int ch = tid + s * 256;
            int r = ch >> 2, cc = (ch & 3) << 3;
            *reinterpret_cast<u32x4*>(&sA[buf][r * LDA + cc]) = ua[s].q;
        }
    };
#endif

    v8f acc[2][4];
    #pragma unroll
    for (int i = 0; i < 2; ++i)
        #pragma unroll
        for (int j = 0; j < 4; ++j) acc[i][j] = zero8();

    const int ksteps = Kdim >> 5;

    // ---- prologue: stage tile 0 into buffer 0 ----
    {
        U4u ub[2];
#if USE_ASYNC_LDS
        stageA(0, 0);
#else
        U4u ua[2];
        loadA(0, ua);
        writeA(0, ua);
#endif
        loadB(0, ub);
        writeB(0, ub);
        ASYNC_WAIT();
        __syncthreads();
    }

    int buf = 0;
    for (int kt = 0; kt < ksteps; ++kt) {
        const bool havenext = (kt + 1 < ksteps);
        const int k0n = (kt + 1) << 5;
        U4u nb[2];
#if !USE_ASYNC_LDS
        U4u na[2];
#endif
        if (havenext) {                      // issue next-tile fetches early
#if USE_ASYNC_LDS
            stageA(k0n, buf ^ 1);
#else
            loadA(k0n, na);
#endif
            loadB(k0n, nb);
            if (kt + 2 < ksteps) {           // warm L2 for tile kt+2
                __builtin_prefetch(&Ag[(size_t)(m0 + (tid >> 2)) * Kdim + k0n + 32], 0, 1);
                __builtin_prefetch(&Bg[(size_t)(k0n + 32 + (tid >> 4)) * Ndim + n0], 0, 1);
            }
        }

        // ---- fragments + WMMA on current buffer ----
        // A frag (ISA 16-bit A layout): lane half selects K sub-runs of 8
        ABu a[2];
        #pragma unroll
        for (int mt = 0; mt < 2; ++mt) {
            int r = wm * 32 + mt * 16 + lr;
            a[mt].q[0] = *reinterpret_cast<const u32x4*>(&sA[buf][r * LDA + half * 8]);
            a[mt].q[1] = *reinterpret_cast<const u32x4*>(&sA[buf][r * LDA + 16 + half * 8]);
        }
        #pragma unroll
        for (int nt = 0; nt < 4; ++nt) {
            // B frag: same K order as A frag (half selects K sub-runs of 8)
            ABu bfr;
            int br = wn * 64 + nt * 16 + lr;
            bfr.q[0] = *reinterpret_cast<const u32x4*>(&sB[buf][br * LDB + half * 8]);
            bfr.q[1] = *reinterpret_cast<const u32x4*>(&sB[buf][br * LDB + 16 + half * 8]);
            #pragma unroll
            for (int mt = 0; mt < 2; ++mt)
                acc[mt][nt] = __builtin_amdgcn_wmma_f32_16x16x32_bf16(
                    false, a[mt].v, false, bfr.v, (short)0, acc[mt][nt], false, false);
        }

        if (havenext) {                      // commit next tile, flip buffers
#if !USE_ASYNC_LDS
            writeA(buf ^ 1, na);
#endif
            writeB(buf ^ 1, nb);
            ASYNC_WAIT();
            __syncthreads();
            buf ^= 1;
        }
    }

    // ---- epilogue (C layout: vgpr j = row half*8+j, lane lr = col) ----
    #pragma unroll
    for (int mt = 0; mt < 2; ++mt) {
        #pragma unroll
        for (int nt = 0; nt < 4; ++nt) {
            #pragma unroll
            for (int jv = 0; jv < 8; ++jv) {
                int grow = m0 + wm * 32 + mt * 16 + half * 8 + jv;
                int gcol = n0 + wn * 64 + nt * 16 + lr;
                size_t idx = (size_t)grow * Ndim + gcol;
                float v = acc[mt][nt][jv] + bias[gcol];
                if constexpr (EPI == 0) {
                    Cb[idx] = f2b(v);
                } else if constexpr (EPI == 1) {
                    Cf[idx] = extra[idx] + v;
                } else if constexpr (EPI == 2) {
                    v = 0.5f * v * (1.0f + erff(v * 0.70710678118654752f));
                    Cb[idx] = f2b(v);
                } else {
                    Cf[idx] = v + extra[(size_t)(grow % Tm) * Ndim + gcol];
                }
            }
        }
    }
}

// =====================================================================
// Bucketed local attention. One block per (batch, bucket, head),
// 128 threads = 4 wave32; each wave owns 16 query rows.
// scores: 64x128 (K=64) -> masked softmax -> out: 64x64 (K=128), all WMMA.
// Q/K staged with async global->LDS when available.
// =====================================================================
__global__ __launch_bounds__(128) void attn_kernel(
    const u16* __restrict__ qkv, u16* __restrict__ attnout)
{
    constexpr int SQS = 72, SKS = 72, SVS = 136, SPS = 136; // padded strides
    __shared__ __align__(16) u16 sQ [64  * SQS];
    __shared__ __align__(16) u16 sK [128 * SKS];   // [key 0..127][d]
    __shared__ __align__(16) u16 sVt[64  * SVS];   // [d][key]
    __shared__ __align__(16) u16 sP [64  * SPS];   // probs, A-layout (row-major)

    const int nb  = Tc / BSc;
    const int blk = blockIdx.x;
    const int b   = blk / (nb * Hc);
    const int rem = blk % (nb * Hc);
    const int bkt = rem / Hc;
    const int h   = rem % Hc;

    const int tid = threadIdx.x;
    const size_t rs3c = 3 * Cc;
    const size_t base = ((size_t)b * Tc) * rs3c + (size_t)h * HDc;

    // ---- stage Q ----
    #pragma unroll
    for (int s = 0; s < 4; ++s) {
        int ch = tid + s * 128;
        int r = ch >> 3, d = (ch & 7) << 3;
        int t = bkt * BSc + r;
#if USE_ASYNC_LDS
        ASYNC_CP16(&qkv[base + (size_t)t * rs3c + d], &sQ[r * SQS + d]);
#else
        *reinterpret_cast<u32x4*>(&sQ[r * SQS + d]) =
            *reinterpret_cast<const u32x4*>(&qkv[base + (size_t)t * rs3c + d]);
#endif
    }
    // ---- stage K = [prev bucket | current bucket] ----
    #pragma unroll
    for (int s = 0; s < 8; ++s) {
        int ch = tid + s * 128;
        int key = ch >> 3, d = (ch & 7) << 3;
        if (key < 64 && bkt == 0) {
            u32x4 z = {0u, 0u, 0u, 0u};
            *reinterpret_cast<u32x4*>(&sK[key * SKS + d]) = z;
        } else {
            int t = (key < 64) ? ((bkt - 1) * BSc + key) : (bkt * BSc + key - 64);
#if USE_ASYNC_LDS
            ASYNC_CP16(&qkv[base + Cc + (size_t)t * rs3c + d], &sK[key * SKS + d]);
#else
            *reinterpret_cast<u32x4*>(&sK[key * SKS + d]) =
                *reinterpret_cast<const u32x4*>(&qkv[base + Cc + (size_t)t * rs3c + d]);
#endif
        }
    }
    // ---- stage V transposed: sVt[d][key] (manual transpose scatter) ----
    #pragma unroll
    for (int s = 0; s < 8; ++s) {
        int ch = tid + s * 128;
        int key = ch >> 3, d = (ch & 7) << 3;
        U4u u;
        u.q = u32x4{0u, 0u, 0u, 0u};
        if (!(key < 64 && bkt == 0)) {
            int t = (key < 64) ? ((bkt - 1) * BSc + key) : (bkt * BSc + key - 64);
            u.q = *reinterpret_cast<const u32x4*>(&qkv[base + 2 * Cc + (size_t)t * rs3c + d]);
        }
        #pragma unroll
        for (int e = 0; e < 8; ++e) sVt[(d + e) * SVS + key] = u.s[e];
    }
    ASYNC_WAIT();
    __syncthreads();

    const int wid = tid >> 5, lane = tid & 31;
    const int lr = lane & 15, half = lane >> 4;
    const int mw = wid * 16;                       // this wave's query rows

    // ---- scores S = Q K^T  (64x128, K=64) ----
    v8f s[8];
    #pragma unroll
    for (int nt = 0; nt < 8; ++nt) s[nt] = zero8();
    #pragma unroll
    for (int kt = 0; kt < 2; ++kt) {
        int k0 = kt * 32;
        ABu a;
        int ar = mw + lr;
        a.q[0] = *reinterpret_cast<const u32x4*>(&sQ[ar * SQS + k0 + half * 8]);
        a.q[1] = *reinterpret_cast<const u32x4*>(&sQ[ar * SQS + k0 + 16 + half * 8]);
        #pragma unroll
        for (int nt = 0; nt < 8; ++nt) {
            ABu bb;
            int br = nt * 16 + lr;
            bb.q[0] = *reinterpret_cast<const u32x4*>(&sK[br * SKS + k0 + half * 8]);
            bb.q[1] = *reinterpret_cast<const u32x4*>(&sK[br * SKS + k0 + 16 + half * 8]);
            s[nt] = __builtin_amdgcn_wmma_f32_16x16x32_bf16(
                false, a.v, false, bb.v, (short)0, s[nt], false, false);
        }
    }

    // ---- masked softmax across 128 cols (row = mw + half*8 + jv) ----
    const float scale = 0.125f;                    // 1/sqrt(64)
    #pragma unroll
    for (int jv = 0; jv < 8; ++jv) {
        int row = mw + half * 8 + jv;
        float mx = -3.0e38f;
        #pragma unroll
        for (int nt = 0; nt < 8; ++nt) {
            int col = nt * 16 + lr;
            float v = s[nt][jv] * scale;
            bool ok = (col < 64) || (col - 64 <= row);
            v = ok ? v : -3.0e38f;
            s[nt][jv] = v;
            mx = fmaxf(mx, v);
        }
        #pragma unroll
        for (int o = 8; o >= 1; o >>= 1) mx = fmaxf(mx, __shfl_xor(mx, o, 32));
        float sm = 0.f;
        #pragma unroll
        for (int nt = 0; nt < 8; ++nt) {
            float e = __expf(s[nt][jv] - mx);
            s[nt][jv] = e;
            sm += e;
        }
        #pragma unroll
        for (int o = 8; o >= 1; o >>= 1) sm += __shfl_xor(sm, o, 32);
        float inv = 1.0f / sm;
        #pragma unroll
        for (int nt = 0; nt < 8; ++nt)
            sP[row * SPS + nt * 16 + lr] = f2b(s[nt][jv] * inv);
    }
    __syncthreads();

    // ---- O = P V  (64x64, K=128) ----
    v8f o[4];
    #pragma unroll
    for (int nt = 0; nt < 4; ++nt) o[nt] = zero8();
    #pragma unroll
    for (int kt = 0; kt < 4; ++kt) {
        int k0 = kt * 32;
        ABu a;
        int ar = mw + lr;
        a.q[0] = *reinterpret_cast<const u32x4*>(&sP[ar * SPS + k0 + half * 8]);
        a.q[1] = *reinterpret_cast<const u32x4*>(&sP[ar * SPS + k0 + 16 + half * 8]);
        #pragma unroll
        for (int nt = 0; nt < 4; ++nt) {
            ABu bb;
            int br = nt * 16 + lr;
            bb.q[0] = *reinterpret_cast<const u32x4*>(&sVt[br * SVS + k0 + half * 8]);
            bb.q[1] = *reinterpret_cast<const u32x4*>(&sVt[br * SVS + k0 + 16 + half * 8]);
            o[nt] = __builtin_amdgcn_wmma_f32_16x16x32_bf16(
                false, a.v, false, bb.v, (short)0, o[nt], false, false);
        }
    }
    #pragma unroll
    for (int nt = 0; nt < 4; ++nt) {
        #pragma unroll
        for (int jv = 0; jv < 8; ++jv) {
            int row = mw + half * 8 + jv;
            int t = bkt * BSc + row;
            int col = nt * 16 + lr;
            attnout[((size_t)(b * Tc + t)) * Cc + h * HDc + col] = f2b(o[nt][jv]);
        }
    }
}

// =====================================================================
// LayerNorm: one block per row (C=1024), bf16 output.
// =====================================================================
__global__ __launch_bounds__(256) void ln_kernel(
    const float* __restrict__ x, const float* __restrict__ w,
    const float* __restrict__ bptr, u16* __restrict__ outp, int Cdim)
{
    int row = blockIdx.x;
    const float* xr = x + (size_t)row * Cdim;
    int tid = threadIdx.x;
    float s = 0.f, sq = 0.f;
    for (int c = tid; c < Cdim; c += 256) { float v = xr[c]; s += v; sq += v * v; }
    #pragma unroll
    for (int o = 16; o >= 1; o >>= 1) { s += __shfl_xor(s, o, 32); sq += __shfl_xor(sq, o, 32); }
    __shared__ float rsm[8], rqm[8];
    int wid = tid >> 5, lane = tid & 31;
    if (lane == 0) { rsm[wid] = s; rqm[wid] = sq; }
    __syncthreads();
    float ts = 0.f, tq = 0.f;
    #pragma unroll
    for (int i = 0; i < 8; ++i) { ts += rsm[i]; tq += rqm[i]; }
    float mean = ts / Cdim;
    float var  = tq / Cdim - mean * mean;
    float rst  = rsqrtf(var + 1e-5f);
    for (int c = tid; c < Cdim; c += 256)
        outp[(size_t)row * Cdim + c] = f2b((xr[c] - mean) * rst * w[c] + bptr[c]);
}

// =====================================================================
// fp32 -> bf16 convert (grid-stride)
// =====================================================================
__global__ void cvt_f32_bf16(const float* __restrict__ in, u16* __restrict__ out, size_t n)
{
    size_t i = (size_t)blockIdx.x * blockDim.x + threadIdx.x;
    size_t stride = (size_t)gridDim.x * blockDim.x;
    for (; i < n; i += stride) out[i] = f2b(in[i]);
}

// =====================================================================
// Actor + critic heads. Block = 64 threads per sequence row.
// out layout: [logits M*64 | logstd 64 | values M]
// =====================================================================
__global__ __launch_bounds__(64) void heads_kernel(
    const u16* __restrict__ xb, const float* __restrict__ aw,
    const float* __restrict__ ab, const float* __restrict__ cw,
    const float* __restrict__ cb, const float* __restrict__ logstd,
    float* __restrict__ out, int M)
{
    int row = blockIdx.x;
    int t = threadIdx.x;
    const u16* xr = xb + (size_t)row * Cc;
    float acc = 0.f, cacc = 0.f;
    for (int k = 0; k < Cc; ++k) acc += b2f(xr[k]) * aw[k * Ac + t];
    for (int k = t; k < Cc; k += 64) cacc += b2f(xr[k]) * cw[k];
    #pragma unroll
    for (int o = 16; o >= 1; o >>= 1) cacc += __shfl_xor(cacc, o, 32);
    __shared__ float cr[2];
    int wid = t >> 5, lane = t & 31;
    if (lane == 0) cr[wid] = cacc;
    __syncthreads();
    out[(size_t)row * Ac + t] = acc + ab[t];
    if (t == 0) out[(size_t)M * Ac + 64 + row] = cr[0] + cr[1] + cb[0];
    if (row == 0) out[(size_t)M * Ac + t] = logstd[t];
}

// =====================================================================
// Orchestration
// =====================================================================
extern "C" void kernel_launch(void* const* d_in, const int* in_sizes, int n_in,
                              void* d_out, int out_size, void* d_ws, size_t ws_size,
                              hipStream_t stream)
{
    (void)in_sizes; (void)n_in; (void)out_size; (void)ws_size;
    const float* states  = (const float*)d_in[0];
    const float* se_w    = (const float*)d_in[1];
    const float* se_b    = (const float*)d_in[2];
    const float* pos     = (const float*)d_in[3];
    const float* ln1_w   = (const float*)d_in[4];
    const float* ln1_b   = (const float*)d_in[5];
    const float* qkv_w   = (const float*)d_in[6];
    const float* qkv_b   = (const float*)d_in[7];
    const float* ao_w    = (const float*)d_in[8];
    const float* ao_b    = (const float*)d_in[9];
    const float* ln2_w   = (const float*)d_in[10];
    const float* ln2_b   = (const float*)d_in[11];
    const float* w1      = (const float*)d_in[12];
    const float* b1      = (const float*)d_in[13];
    const float* w2      = (const float*)d_in[14];
    const float* b2      = (const float*)d_in[15];
    const float* lnf_w   = (const float*)d_in[16];
    const float* lnf_b   = (const float*)d_in[17];
    const float* actor_w = (const float*)d_in[18];
    const float* actor_b = (const float*)d_in[19];
    const float* logstd  = (const float*)d_in[20];
    const float* crit_w  = (const float*)d_in[21];
    const float* crit_b  = (const float*)d_in[22];
    float* out = (float*)d_out;

    // bump-allocate workspace
    char* p = (char*)d_ws;
    auto take = [&](size_t bytes) -> char* {
        char* r = p;
        p += (bytes + 255) & ~(size_t)255;
        return r;
    };
    const int M = Mc;
    float* xf  = (float*)take((size_t)M * Cc * 4);      // residual stream (f32)
    u16* hb    = (u16*)take((size_t)M * Cc * 2);        // LN output (bf16)
    u16* qkvb  = (u16*)take((size_t)M * 3 * Cc * 2);    // QKV (bf16)
    u16* attnb = (u16*)take((size_t)M * Cc * 2);        // attn output (bf16)
    u16* hidb  = (u16*)take((size_t)M * 4 * Cc * 2);    // MLP hidden (bf16)
    u16* st_b  = (u16*)take((size_t)M * SDc * 2);
    u16* sew_b = (u16*)take((size_t)SDc * Cc * 2);
    u16* qw_b  = (u16*)take((size_t)Lc * Cc * 3 * Cc * 2);
    u16* aw_b  = (u16*)take((size_t)Lc * Cc * Cc * 2);
    u16* w1_b  = (u16*)take((size_t)Lc * Cc * 4 * Cc * 2);
    u16* w2_b  = (u16*)take((size_t)Lc * 4 * Cc * Cc * 2);

    // convert activations + weights to bf16
    cvt_f32_bf16<<<4096, 256, 0, stream>>>(states, st_b, (size_t)M * SDc);
    cvt_f32_bf16<<<256, 256, 0, stream>>>(se_w, sew_b, (size_t)SDc * Cc);
    cvt_f32_bf16<<<8192, 256, 0, stream>>>(qkv_w, qw_b, (size_t)Lc * Cc * 3 * Cc);
    cvt_f32_bf16<<<4096, 256, 0, stream>>>(ao_w, aw_b, (size_t)Lc * Cc * Cc);
    cvt_f32_bf16<<<8192, 256, 0, stream>>>(w1, w1_b, (size_t)Lc * Cc * 4 * Cc);
    cvt_f32_bf16<<<8192, 256, 0, stream>>>(w2, w2_b, (size_t)Lc * 4 * Cc * Cc);

    // x = states @ W_embed + b + pos
    gemm_bf16<3><<<dim3(Cc / 128, M / 128), 256, 0, stream>>>(
        st_b, sew_b, se_b, pos, xf, nullptr, M, Cc, SDc, Tc);

    for (int l = 0; l < Lc; ++l) {
        ln_kernel<<<M, 256, 0, stream>>>(xf, ln1_w + l * Cc, ln1_b + l * Cc, hb, Cc);
        gemm_bf16<0><<<dim3(3 * Cc / 128, M / 128), 256, 0, stream>>>(
            hb, qw_b + (size_t)l * Cc * 3 * Cc, qkv_b + l * 3 * Cc,
            nullptr, nullptr, qkvb, M, 3 * Cc, Cc, Tc);
        attn_kernel<<<Bc * (Tc / BSc) * Hc, 128, 0, stream>>>(qkvb, attnb);
        gemm_bf16<1><<<dim3(Cc / 128, M / 128), 256, 0, stream>>>(
            attnb, aw_b + (size_t)l * Cc * Cc, ao_b + l * Cc,
            xf, xf, nullptr, M, Cc, Cc, Tc);
        ln_kernel<<<M, 256, 0, stream>>>(xf, ln2_w + l * Cc, ln2_b + l * Cc, hb, Cc);
        gemm_bf16<2><<<dim3(4 * Cc / 128, M / 128), 256, 0, stream>>>(
            hb, w1_b + (size_t)l * Cc * 4 * Cc, b1 + l * 4 * Cc,
            nullptr, nullptr, hidb, M, 4 * Cc, Cc, Tc);
        gemm_bf16<1><<<dim3(Cc / 128, M / 128), 256, 0, stream>>>(
            hidb, w2_b + (size_t)l * 4 * Cc * Cc, b2 + l * Cc,
            xf, xf, nullptr, M, Cc, 4 * Cc, Tc);
    }

    ln_kernel<<<M, 256, 0, stream>>>(xf, lnf_w, lnf_b, hb, Cc);
    heads_kernel<<<M, 64, 0, stream>>>(hb, actor_w, actor_b, crit_w, crit_b,
                                       logstd, out, M);
}
